// Model_19327352832541
// MI455X (gfx1250) — compile-verified, matrix operand order checked by reading, not run on hardware
//
#include <hip/hip_runtime.h>

typedef __attribute__((ext_vector_type(16))) __bf16 v16bf;
typedef __attribute__((ext_vector_type(8)))  float  v8f;

union U16 { v16bf v; unsigned short h[16]; float4 q[2]; };

__device__ __forceinline__ unsigned short f2bf(float f) {
    unsigned u = __float_as_uint(f);
    u += 0x7FFFu + ((u >> 16) & 1u);          // round-to-nearest-even
    return (unsigned short)(u >> 16);
}

// One WMMA fragment: 16 bf16 at offsets {0..7} and {16..23} from p -> two b128 loads.
__device__ __forceinline__ v16bf ldfrag(const unsigned short* __restrict__ p) {
    U16 u;
    u.q[0] = *(const float4*)(p);
    u.q[1] = *(const float4*)(p + 16);
    return u.v;
}

// C[M,N] (f32) = op(A[M,K] @ B^T + bias); A,B bf16 row-major, B given as [N,K].
// Requirements: M % 64 == 0, N % 16 == 0, K % 32 == 0 (caller pads with zeros).
// One wave computes 4 m-tiles x 1 n-tile, sharing the B fragment.
__global__ void wmma_gemm_bf16(const unsigned short* __restrict__ A, int lda,
                               const unsigned short* __restrict__ B, int ldb,
                               float* __restrict__ C, int ldc,
                               int mGroups, int tilesN, int K,
                               const float* __restrict__ bias, int act)
{
    const int lane = threadIdx.x & 31;
    const int wid  = blockIdx.x * (blockDim.x >> 5) + (threadIdx.x >> 5);
    if (wid >= mGroups * tilesN) return;            // wave-uniform
    const int m0 = (wid / tilesN) << 6;             // 64 rows per wave
    const int n0 = (wid % tilesN) << 4;
    const int lane15   = lane & 15;
    const int laneHalf = lane >> 4;
    const int kbase    = laneHalf << 3;

    const unsigned short* b  = B + (size_t)(n0 + lane15) * ldb + kbase;
    const unsigned short* a0 = A + (size_t)(m0 + lane15) * lda + kbase;
    const unsigned short* a1 = a0 + (size_t)16 * lda;
    const unsigned short* a2 = a0 + (size_t)32 * lda;
    const unsigned short* a3 = a0 + (size_t)48 * lda;

    v8f c0 = {}, c1 = {}, c2 = {}, c3 = {};
    for (int k0 = 0; k0 < K; k0 += 32) {
        v16bf fb = ldfrag(b  + k0);
        v16bf f0 = ldfrag(a0 + k0);
        v16bf f1 = ldfrag(a1 + k0);
        v16bf f2 = ldfrag(a2 + k0);
        v16bf f3 = ldfrag(a3 + k0);
        c0 = __builtin_amdgcn_wmma_f32_16x16x32_bf16(false, f0, false, fb, (short)0, c0, false, false);
        c1 = __builtin_amdgcn_wmma_f32_16x16x32_bf16(false, f1, false, fb, (short)0, c1, false, false);
        c2 = __builtin_amdgcn_wmma_f32_16x16x32_bf16(false, f2, false, fb, (short)0, c2, false, false);
        c3 = __builtin_amdgcn_wmma_f32_16x16x32_bf16(false, f3, false, fb, (short)0, c3, false, false);
    }

    const int cn = n0 + lane15;
    const float bv = bias ? bias[cn] : 0.0f;
#define WRITE_TILE(cc, toff)                                                   \
    {                                                                          \
        _Pragma("unroll")                                                      \
        for (int v = 0; v < 8; ++v) {                                          \
            float r = cc[v] + bv;                                              \
            if (act) r = tanhf(r);                                             \
            C[(size_t)(m0 + toff + v + (laneHalf << 3)) * ldc + cn] = r;       \
        }                                                                      \
    }
    WRITE_TILE(c0, 0) WRITE_TILE(c1, 16) WRITE_TILE(c2, 32) WRITE_TILE(c3, 48)
#undef WRITE_TILE
}

// ---- elementwise f32 -> bf16 ----
__global__ void cvt_bf16_kernel(const float* __restrict__ src,
                                unsigned short* __restrict__ dst, int n) {
    int i = blockIdx.x * blockDim.x + threadIdx.x;
    if (i < n) dst[i] = f2bf(src[i]);
}

// enc_seq f32 [>=257,512] -> bf16 [288,512], rows >= validRows zeroed
__global__ void cvt_pad_rows_kernel(const float* __restrict__ src,
                                    unsigned short* __restrict__ dst, int validRows) {
    int i = blockIdx.x * blockDim.x + threadIdx.x;     // 288*512
    int r = i >> 9;
    dst[i] = (r < validRows) ? f2bf(src[i]) : (unsigned short)0;
}

// h_seq f32 [>=257,512] -> h_seq^T bf16 [512,288], cols >= validRows zeroed
__global__ void transpose_bf16_kernel(const float* __restrict__ src,
                                      unsigned short* __restrict__ dst, int validRows) {
    int i = blockIdx.x * blockDim.x + threadIdx.x;     // 512*288
    int j = i / 288, s = i % 288;
    dst[i] = (s < validRows) ? f2bf(src[(size_t)s * 512 + j]) : (unsigned short)0;
}

// ---- embeddings: gather + L2-normalize (eps 1e-12), emit bf16 ----
__global__ void embed_norm_kernel(const int* __restrict__ src_nums,
                                  const int* __restrict__ tgt_nums,
                                  const float* __restrict__ enc_emb,
                                  const float* __restrict__ dec_emb,
                                  unsigned short* __restrict__ E_src,
                                  unsigned short* __restrict__ E_dec)
{
    int bidx = blockIdx.x;                 // 0..511
    const float* row; unsigned short* outp;
    if (bidx < 256) { row = enc_emb + (size_t)src_nums[bidx] * 512; outp = E_src + (size_t)bidx * 512; }
    else { int t = bidx - 256; row = dec_emb + (size_t)tgt_nums[t] * 512; outp = E_dec + (size_t)t * 512; }
    __shared__ float red[256];
    int tid = threadIdx.x;                 // 256 threads
    float x0 = row[tid], x1 = row[tid + 256];
    red[tid] = x0 * x0 + x1 * x1;
    __syncthreads();
    for (int s = 128; s > 0; s >>= 1) { if (tid < s) red[tid] += red[tid + s]; __syncthreads(); }
    float n = sqrtf(red[0]);
    float d = 1.0f / fmaxf(n, 1e-12f);
    outp[tid]       = f2bf(x0 * d);
    outp[tid + 256] = f2bf(x1 * d);
}

// ---- device-scope split barrier (generation counter) ----
__device__ __forceinline__ void gbar(int* cnt, volatile int* gen, int nb, int target) {
    __syncthreads();
    if (threadIdx.x == 0) {
        __threadfence();
        if (atomicAdd(cnt, 1) == nb - 1) {
            atomicExch(cnt, 0);
            __threadfence();
            atomicAdd((int*)gen, 1);
        }
        while (*gen < target) __builtin_amdgcn_s_sleep(1);
        __threadfence();
    }
    __syncthreads();
}

__global__ void reset_barrier_kernel(int* bar) { if (threadIdx.x < 4) bar[threadIdx.x] = 0; }

// ---- sequential RNNs: seq[t+1] = tanh(P[t] + W_hh @ seq[t]) ----
#define RNN_BLOCKS 32
#define ROWS_PER_BLK 16
__global__ void rnn_seq_kernel(const float* __restrict__ P_enc, const float* __restrict__ Whh_enc,
                               const float* __restrict__ h0_enc, float* __restrict__ seq_enc,
                               const float* __restrict__ P_dec, const float* __restrict__ Whh_dec,
                               const float* __restrict__ h0_dec, float* __restrict__ seq_dec,
                               int T, int* __restrict__ bar)
{
    const int grp = blockIdx.x / RNN_BLOCKS;       // 0 = encoder, 1 = decoder
    const int blk = blockIdx.x % RNN_BLOCKS;
    const float* P   = grp ? P_dec   : P_enc;
    const float* Whh = grp ? Whh_dec : Whh_enc;
    const float* h0  = grp ? h0_dec  : h0_enc;
    volatile float* seq = grp ? seq_dec : seq_enc;
    int* cnt          = bar + grp * 2;
    volatile int* gen = bar + grp * 2 + 1;

    __shared__ float Wsl[ROWS_PER_BLK * 512];
    __shared__ float hbuf[512];

    const int tid = threadIdx.x;                   // 512 threads = 16 waves
    const int rowbase = blk * ROWS_PER_BLK;
    for (int idx = tid; idx < ROWS_PER_BLK * 512; idx += 512)
        Wsl[idx] = Whh[(size_t)rowbase * 512 + idx];
    if (blk == 0) seq[tid] = h0[tid];              // row 0 = h0
    gbar(cnt, gen, RNN_BLOCKS, 1);

    const int wv = tid >> 5, ln = tid & 31;        // wave wv computes row rowbase+wv
    for (int t = 0; t < T; ++t) {
        hbuf[tid] = seq[(size_t)t * 512 + tid];
        __syncthreads();
        float acc = 0.0f;
        const float* wr = &Wsl[wv * 512];
        for (int c = ln; c < 512; c += 32)
            acc += wr[c] * hbuf[c];
        for (int off = 16; off > 0; off >>= 1)
            acc += __shfl_down(acc, off, 32);
        if (ln == 0) {
            int grow = rowbase + wv;
            seq[(size_t)(t + 1) * 512 + grow] = tanhf(acc + P[(size_t)t * 512 + grow]);
        }
        gbar(cnt, gen, RNN_BLOCKS, t + 2);
    }
}

// ---- copy decoder states into CH[:,512:1024] (f32) ----
__global__ void copy_hdec_kernel(const float* __restrict__ dec_seq, float* __restrict__ CH) {
    int idx = blockIdx.x * blockDim.x + threadIdx.x;   // 256*512 total
    int t = idx >> 9, j = idx & 511;
    CH[(size_t)t * 1024 + 512 + j] = dec_seq[idx];
}

// ---- softmax over 257 valid cols of f32 scores [256,288]; emit bf16 weights [256,288] ----
__global__ void softmax_row_kernel(const float* __restrict__ S, unsigned short* __restrict__ Wb) {
    const float* row = S + (size_t)blockIdx.x * 288;
    unsigned short* wrow = Wb + (size_t)blockIdx.x * 288;
    __shared__ float red[256];
    int tid = threadIdx.x;
    float v0 = row[tid];
    float v1 = (tid == 0) ? row[256] : -__builtin_inff();
    red[tid] = fmaxf(v0, v1);
    __syncthreads();
    for (int s = 128; s > 0; s >>= 1) { if (tid < s) red[tid] = fmaxf(red[tid], red[tid + s]); __syncthreads(); }
    float mx = red[0];
    __syncthreads();
    float e0 = expf(v0 - mx);
    float e1 = (tid == 0) ? expf(v1 - mx) : 0.0f;
    red[tid] = e0 + e1;
    __syncthreads();
    for (int s = 128; s > 0; s >>= 1) { if (tid < s) red[tid] += red[tid + s]; __syncthreads(); }
    float inv = 1.0f / red[0];
    wrow[tid] = f2bf(e0 * inv);
    if (tid == 0) wrow[256] = f2bf(e1 * inv);
    if (tid < 31) wrow[257 + tid] = 0;             // zero padding cols
}

// ---- per-row logsumexp + NLL ----
__global__ void nll_kernel(const float* __restrict__ logits, const int* __restrict__ tgt,
                           float* __restrict__ nll) {
    int t = blockIdx.x;
    const float* row = logits + (size_t)t * 32000;
    __shared__ float red[256];
    int tid = threadIdx.x;
    float mx = -__builtin_inff();
    for (int j = tid; j < 32000; j += 256) mx = fmaxf(mx, row[j]);
    red[tid] = mx; __syncthreads();
    for (int s = 128; s > 0; s >>= 1) { if (tid < s) red[tid] = fmaxf(red[tid], red[tid + s]); __syncthreads(); }
    mx = red[0];
    __syncthreads();
    float sum = 0.0f;
    for (int j = tid; j < 32000; j += 256) sum += expf(row[j] - mx);
    red[tid] = sum; __syncthreads();
    for (int s = 128; s > 0; s >>= 1) { if (tid < s) red[tid] += red[tid + s]; __syncthreads(); }
    if (tid == 0) nll[t] = (mx + logf(red[0])) - row[tgt[t]];
}

__global__ void mean_kernel(const float* __restrict__ nll, float* __restrict__ out) {
    __shared__ float red[256];
    int tid = threadIdx.x;
    red[tid] = nll[tid];
    __syncthreads();
    for (int s = 128; s > 0; s >>= 1) { if (tid < s) red[tid] += red[tid + s]; __syncthreads(); }
    if (tid == 0) out[0] = red[0] * (1.0f / 256.0f);
}

extern "C" void kernel_launch(void* const* d_in, const int* in_sizes, int n_in,
                              void* d_out, int out_size, void* d_ws, size_t ws_size,
                              hipStream_t stream)
{
    const int*   src_nums = (const int*)d_in[0];
    const int*   tgt_nums = (const int*)d_in[1];
    const float* enc_emb  = (const float*)d_in[2];
    const float* enc_h0   = (const float*)d_in[3];
    const float* enc_W_hi = (const float*)d_in[4];
    const float* enc_W_hh = (const float*)d_in[5];
    const float* enc_b    = (const float*)d_in[6];
    const float* dec_emb  = (const float*)d_in[7];
    const float* dec_h0   = (const float*)d_in[8];
    const float* dec_W_hi = (const float*)d_in[9];
    const float* dec_W_hh = (const float*)d_in[10];
    const float* dec_b    = (const float*)d_in[11];
    const float* tnh_W    = (const float*)d_in[12];
    const float* tnh_b    = (const float*)d_in[13];
    const float* out_W    = (const float*)d_in[14];
    float* out = (float*)d_out;

    char* ws = (char*)d_ws;
    size_t off = 0;
    auto allocB = [&](size_t bytes) -> void* {
        void* p = (void*)(ws + off);
        off += ((bytes + 255) & ~(size_t)255);
        return p;
    };
    auto allocF = [&](size_t n) -> float*          { return (float*)allocB(n * 4); };
    auto allocH = [&](size_t n) -> unsigned short* { return (unsigned short*)allocB(n * 2); };

    // f32 scratch
    float* P_enc   = allocF(256 * 512);
    float* P_dec   = allocF(256 * 512);
    float* enc_seq = allocF(272 * 512);                 // rows 0..256 used
    float* dec_seq = allocF(272 * 512);                 // rows 0..255 used
    float* CH      = allocF(256 * 1024);                // [ctx | h]
    float* scores  = allocF(256 * 288);
    float* Z       = allocF(256 * 512);
    float* logits  = allocF((size_t)256 * 32000);
    float* nll     = allocF(256);
    int*   bar     = (int*)allocF(64);
    // bf16 scratch
    unsigned short* E_src_b   = allocH(256 * 512);
    unsigned short* E_dec_b   = allocH(256 * 512);
    unsigned short* Whi_enc_b = allocH(512 * 512);
    unsigned short* Whi_dec_b = allocH(512 * 512);
    unsigned short* tnhW_b    = allocH(512 * 1024);
    unsigned short* outW_b    = allocH((size_t)32000 * 512);
    unsigned short* dec_seq_b = allocH(256 * 512);
    unsigned short* enc_seq_b = allocH(288 * 512);      // zero-padded rows
    unsigned short* hseqT_b   = allocH(512 * 288);      // zero-padded cols
    unsigned short* weights_b = allocH(256 * 288);      // zero-padded cols
    unsigned short* CH_b      = allocH(256 * 1024);
    unsigned short* Z_b       = allocH(256 * 512);

    auto cvt = [&](const float* s, unsigned short* d, size_t n) {
        cvt_bf16_kernel<<<(int)((n + 255) / 256), 256, 0, stream>>>(s, d, (int)n);
    };
    auto gemm = [&](const unsigned short* A, int lda, const unsigned short* B, int ldb,
                    float* C, int ldc, int M, int N, int K, const float* bias, int act) {
        int mGroups = M / 64, tilesN = N / 16;
        int waves = mGroups * tilesN;
        wmma_gemm_bf16<<<(waves + 3) / 4, 128, 0, stream>>>(A, lda, B, ldb, C, ldc,
                                                            mGroups, tilesN, K, bias, act);
    };

    // 1) embeddings (bf16)
    embed_norm_kernel<<<512, 256, 0, stream>>>(src_nums, tgt_nums, enc_emb, dec_emb, E_src_b, E_dec_b);
    // 2) weight conversions
    cvt(enc_W_hi, Whi_enc_b, 512 * 512);
    cvt(dec_W_hi, Whi_dec_b, 512 * 512);
    cvt(tnh_W,    tnhW_b,    512 * 1024);
    cvt(out_W,    outW_b,    (size_t)32000 * 512);
    // 3) hoisted RNN input projections (+bias)
    gemm(E_src_b, 512, Whi_enc_b, 512, P_enc, 512, 256, 512, 512, enc_b, 0);
    gemm(E_dec_b, 512, Whi_dec_b, 512, P_dec, 512, 256, 512, 512, dec_b, 0);
    // 4) both sequential RNNs (multi-WGP, LDS-resident W_hh slices)
    reset_barrier_kernel<<<1, 64, 0, stream>>>(bar);
    rnn_seq_kernel<<<2 * RNN_BLOCKS, 512, 0, stream>>>(P_enc, enc_W_hh, enc_h0, enc_seq,
                                                       P_dec, dec_W_hh, dec_h0, dec_seq,
                                                       256, bar);
    // 5) state conversions / padding / transpose + CH[:,512:]
    copy_hdec_kernel<<<512, 256, 0, stream>>>(dec_seq, CH);
    cvt(dec_seq, dec_seq_b, 256 * 512);
    cvt_pad_rows_kernel<<<576, 256, 0, stream>>>(enc_seq, enc_seq_b, 257);
    transpose_bf16_kernel<<<576, 256, 0, stream>>>(enc_seq, hseqT_b, 257);
    // 6) attention scores[t,s] = <h_t, hseq_s>   (padded cols give exact zeros)
    gemm(dec_seq_b, 512, enc_seq_b, 512, scores, 288, 256, 288, 512, nullptr, 0);
    // 7) softmax over s -> bf16 weights (zero padding)
    softmax_row_kernel<<<256, 256, 0, stream>>>(scores, weights_b);
    // 8) contexts -> CH[:,0:512]
    gemm(weights_b, 288, hseqT_b, 288, CH, 1024, 256, 512, 288, nullptr, 0);
    // 9) Z = tanh(CH @ tnh_W^T + tnh_b)
    cvt(CH, CH_b, 256 * 1024);
    gemm(CH_b, 1024, tnhW_b, 1024, Z, 512, 256, 512, 1024, tnh_b, 1);
    cvt(Z, Z_b, 256 * 512);
    // 10) logits = Z @ out_W^T   (dominant GEMM, 8.4 GFLOP, memory-bound)
    gemm(Z_b, 512, outW_b, 512, logits, 32000, 256, 32000, 512, nullptr, 0);
    // 11) NLL + mean
    nll_kernel<<<256, 256, 0, stream>>>(logits, tgt_nums, nll);
    mean_kernel<<<1, 256, 0, stream>>>(nll, out);
}
